// ReprogrammingLayer_Gary_65455301591210
// MI455X (gfx1250) — matrix operand order repaired and checked by reading, not compiled, not run on hardware
//
#include <hip/hip_runtime.h>
#include <hip/hip_bf16.h>

// ---------------------------------------------------------------------------
// CDNA5 (gfx1250) wave32 WMMA implementation of the ReprogrammingLayer.
// f16 end-to-end: weights pre-transposed/converted once, intermediates f16,
// v_wmma_f32_16x16x32_f16 everywhere, double-buffered LDS, b128 traffic.
// ---------------------------------------------------------------------------

typedef __attribute__((ext_vector_type(16))) _Float16 v16h;
typedef __attribute__((ext_vector_type(8)))  _Float16 h8;
typedef __attribute__((ext_vector_type(8)))  float    v8f;
typedef __attribute__((ext_vector_type(4)))  float    v4f;

__device__ static inline v8f wmma_f16(v16h a, v16h b, v8f c) {
  return __builtin_amdgcn_wmma_f32_16x16x32_f16(
      false, a, false, b, (short)0, c, false, false);
}

__device__ static inline v16h cat16(h8 lo, h8 hi) {
  return __builtin_shufflevector(lo, hi, 0, 1, 2, 3, 4, 5, 6, 7,
                                         8, 9, 10, 11, 12, 13, 14, 15);
}

__device__ static inline h8 cvt8(v4f a, v4f b) {
  h8 r;
#pragma unroll
  for (int i = 0; i < 4; ++i) { r[i] = (_Float16)a[i]; r[4 + i] = (_Float16)b[i]; }
  return r;
}

// ---------------------------------------------------------------------------
// One-time weight conversion: W[K][N] f32 -> WT[N][K] f16 (k-contiguous).
// Reads coalesced (row-major); each thread writes one 16B chunk of a WT row.
// ---------------------------------------------------------------------------
template <int K, int N>
__global__ __launch_bounds__(256) void transpose_w_f16(
    const float* __restrict__ W, _Float16* __restrict__ WT)
{
  const int idx = blockIdx.x * 256 + threadIdx.x;   // N*K/8 threads
  const int n   = idx % N;                          // consecutive lanes -> consecutive n
  const int k8  = (idx / N) * 8;
  h8 o;
#pragma unroll
  for (int i = 0; i < 8; ++i)
    o[i] = (_Float16)W[(size_t)(k8 + i) * N + n];   // coalesced across lanes
  *(h8*)&WT[(size_t)n * K + k8] = o;
}

#define GB_BM 64
#define GB_BN 128
#define GB_BK 32
#define AS_ST 40
#define BS_ST 40

// C[M,N] = scale * (A[M,K] @ W[K,N] + bias[N]);  W given as WT[N][K] f16.
// AF16: A matrix is f16; OUTF16: write C as f16.
template <int N, int K, bool AF16, bool OUTF16>
__global__ __launch_bounds__(256) void gemm_bias_wmma(
    const void* __restrict__ Ap, const _Float16* __restrict__ WT,
    const float* __restrict__ bias, void* __restrict__ Cp,
    int M, const float* sA, const float* sB, float scaleConst)
{
  __shared__ _Float16 As[2][GB_BM][AS_ST];    // A tile [m][k]
  __shared__ _Float16 Bst[2][GB_BN][BS_ST];   // B tile TRANSPOSED [n][k]

  const int tid  = threadIdx.x;
  const int lane = tid & 31;
  const int wid  = tid >> 5;
  const int m0 = blockIdx.x * GB_BM;
  const int n0 = blockIdx.y * GB_BN;
  const int wm = (wid & 1) * 32;
  const int wn = (wid >> 1) * 32;
  const int mrow = lane & 15;
  const int hb8  = (lane >> 4) << 3;
  const int koff = (lane >> 4) << 4;

  const int ar  = tid >> 2;
  const int ac  = (tid & 3) * 8;
  const int arow = (m0 + ar < M) ? (m0 + ar) : (M - 1);   // clamp, no branch
  const int bn  = tid & 127;
  const int bkh = (tid >> 7) * 16;

  const float*    Af32 = (const float*)Ap;
  const _Float16* Af16 = (const _Float16*)Ap;
  const _Float16* WTbase = WT + (size_t)(n0 + bn) * K + bkh;

  auto loadA = [&](int k0) -> h8 {
    if constexpr (AF16) {
      return *(const h8*)(Af16 + (size_t)arow * K + ac + k0);
    } else {
      const float* ap = Af32 + (size_t)arow * K + ac + k0;
      return cvt8(*(const v4f*)ap, *(const v4f*)(ap + 4));
    }
  };
  auto loadB = [&](int k0, h8& o0, h8& o1) {
    const _Float16* wp = WTbase + k0;
    o0 = *(const h8*)wp;
    o1 = *(const h8*)(wp + 8);
  };

  float scl = scaleConst;
  if (sA) scl *= *sA;
  if (sB) scl *= *sB;

  v8f acc[2][2];
  acc[0][0] = {}; acc[0][1] = {}; acc[1][0] = {}; acc[1][1] = {};

  {
    h8 av = loadA(0);
    h8 b0, b1; loadB(0, b0, b1);
    *(h8*)&As[0][ar][ac] = av;
    *(h8*)&Bst[0][bn][bkh]     = b0;
    *(h8*)&Bst[0][bn][bkh + 8] = b1;
  }
  __syncthreads();

  constexpr int TILES = K / GB_BK;
  for (int t = 0; t < TILES; ++t) {
    const int cur = t & 1, nxt = cur ^ 1;
    const bool more = (t + 1 < TILES);

    h8 nav, nb0, nb1;
    if (more) { nav = loadA((t + 1) * GB_BK); loadB((t + 1) * GB_BK, nb0, nb1); }

    v16h af[2], bf[2];
#pragma unroll
    for (int q = 0; q < 2; ++q) {
      int mr = wm + q * 16 + mrow;
      af[q] = cat16(*(const h8*)&As[cur][mr][hb8], *(const h8*)&As[cur][mr][hb8 + 16]);
    }
#pragma unroll
    for (int q = 0; q < 2; ++q) {
      int nc = wn + q * 16 + mrow;
      bf[q] = cat16(*(const h8*)&Bst[cur][nc][hb8], *(const h8*)&Bst[cur][nc][hb8 + 16]);
    }
#pragma unroll
    for (int tm = 0; tm < 2; ++tm)
#pragma unroll
      for (int tn = 0; tn < 2; ++tn)
        acc[tm][tn] = wmma_f16(af[tm], bf[tn], acc[tm][tn]);

    if (more) {
      *(h8*)&As[nxt][ar][ac] = nav;
      *(h8*)&Bst[nxt][bn][bkh]     = nb0;
      *(h8*)&Bst[nxt][bn][bkh + 8] = nb1;
    }
    __syncthreads();
  }

#pragma unroll
  for (int tm = 0; tm < 2; ++tm)
#pragma unroll
    for (int tn = 0; tn < 2; ++tn) {
      int col = n0 + wn + tn * 16 + mrow;
      float bv = bias ? bias[col] : 0.f;
#pragma unroll
      for (int j = 0; j < 8; ++j) {
        int row = m0 + wm + tm * 16 + j + hb8;
        if (row < M) {
          float val = scl * (acc[tm][tn][j] + bv);
          if constexpr (OUTF16) ((_Float16*)Cp)[(size_t)row * N + col] = (_Float16)val;
          else                  ((float*)Cp)[(size_t)row * N + col] = val;
        }
      }
    }
}

// ---------------------------------------------------------------------------
// Flash attention, all-f16 inputs/outputs, softmax in A-layout.
// ---------------------------------------------------------------------------
#define AT_L   1024
#define AT_S   1000
#define AT_HD  1024
#define AT_NEG (-1.0e30f)
#define KS_ST  72
#define VS_ST  40
#define PS_ST  36
#define AT_IT  32

__global__ __launch_bounds__(256) void attn_wmma(
    const _Float16* __restrict__ Q,
    const _Float16* __restrict__ Kb,
    const _Float16* __restrict__ Vb,
    _Float16* __restrict__ O)
{
  __shared__ _Float16 Ks[2][32][KS_ST];      // K tile   [s][e]
  __shared__ _Float16 Vst[2][64][VS_ST];     // V tile TRANSPOSED [e][s]
  __shared__ float    Psc[8][16][PS_ST];     // per-wave raw scores [m][k]
  __shared__ float    Crw[8][16];            // per-wave per-row broadcast slot

  const int tid  = threadIdx.x;
  const int lane = tid & 31;
  const int wid  = tid >> 5;
  const int b = blockIdx.x >> 4;
  const int h = blockIdx.x & 15;
  const int l0 = blockIdx.y * 128 + wid * 16;
  const int mrow = lane & 15;
  const int hb8  = (lane >> 4) << 3;
  const int koff = (lane >> 4) << 4;

  const int ksl = tid >> 3;
  const int ke8 = (tid & 7) * 8;
  const int ve  = tid & 63;
  const int vs8 = (tid >> 6) * 8;

  auto loadK = [&](int s0) -> h8 {
    int sg = s0 + ksl; if (sg >= AT_S) sg = AT_S - 1;        // clamp
    return *(const h8*)(Kb + (size_t)sg * AT_HD + h * 64 + ke8);
  };
  auto loadV = [&](int s0) -> h8 {
    h8 vv;
#pragma unroll
    for (int ss = 0; ss < 8; ++ss) {
      int sg = s0 + vs8 + ss; if (sg >= AT_S) sg = AT_S - 1; // clamp
      vv[ss] = Vb[(size_t)sg * AT_HD + h * 64 + ve];
    }
    return vv;
  };

  // ---- Q fragments: direct f16 b128 loads
  v16h qf[2];
  {
    const _Float16* qp = Q + ((size_t)(b * AT_L + l0 + mrow)) * AT_HD + h * 64;
#pragma unroll
    for (int half = 0; half < 2; ++half) {
      const _Float16* qlo = qp + half * 32 + hb8;
      qf[half] = cat16(*(const h8*)qlo, *(const h8*)(qlo + 16));
    }
  }

  float mI = AT_NEG, lI = 0.f;   // per-lane state for row `mrow` (dup per half)
  v8f ao[4];
#pragma unroll
  for (int nt = 0; nt < 4; ++nt) ao[nt] = {};

  {
    h8 k0v = loadK(0), v0v = loadV(0);
    *(h8*)&Ks[0][ksl][ke8] = k0v;
    *(h8*)&Vst[0][ve][vs8] = v0v;
  }
  __syncthreads();

  for (int it = 0; it < AT_IT; ++it) {
    const int s0 = it * 32;
    const int cur = it & 1, nxt = cur ^ 1;
    const bool more = (it + 1 < AT_IT);

    h8 kn, vn;
    if (more) { kn = loadK(s0 + 32); vn = loadV(s0 + 32); }

    // ---- scores: two 16x16 s-subtiles, K-dim 64 = 2 WMMAs each
    v8f sc[2];
#pragma unroll
    for (int st = 0; st < 2; ++st) {
      v8f a = {};
      int srow = st * 16 + mrow;
#pragma unroll
      for (int half = 0; half < 2; ++half) {
        int base = half * 32 + hb8;
        v16h kfrag = cat16(*(const h8*)&Ks[cur][srow][base],
                           *(const h8*)&Ks[cur][srow][base + 16]);
        a = wmma_f16(qf[half], kfrag, a);
      }
      sc[st] = a;
    }

    // ---- transpose scores C-layout -> A-layout via per-wave LDS
#pragma unroll
    for (int st = 0; st < 2; ++st)
#pragma unroll
      for (int j = 0; j < 8; ++j)
        Psc[wid][j + hb8][st * 16 + mrow] = sc[st][j];

    v4f c0a = *(const v4f*)&Psc[wid][mrow][hb8];
    v4f c0b = *(const v4f*)&Psc[wid][mrow][hb8 + 4];
    v4f c1a = *(const v4f*)&Psc[wid][mrow][16 + hb8];
    v4f c1b = *(const v4f*)&Psc[wid][mrow][16 + hb8 + 4];
    float v[16];
#pragma unroll
    for (int i = 0; i < 4; ++i) {
      v[i] = c0a[i]; v[4 + i] = c0b[i]; v[8 + i] = c1a[i]; v[12 + i] = c1b[i];
    }
    if (s0 + 32 > AT_S) {                    // mask only on the partial tile
#pragma unroll
      for (int i = 0; i < 8; ++i) {
        if (s0 + hb8 + i >= AT_S)      v[i]     = AT_NEG;
        if (s0 + 16 + hb8 + i >= AT_S) v[8 + i] = AT_NEG;
      }
    }

    // ---- per-lane online softmax (one cross-half exchange each)
    float t = v[0];
#pragma unroll
    for (int i = 1; i < 16; ++i) t = fmaxf(t, v[i]);
    t = fmaxf(t, __shfl_xor(t, 16, 32));
    float mNew = fmaxf(mI, t);
    float corr = __expf(mI - mNew);
    float p[16];
    float rs = 0.f;
#pragma unroll
    for (int i = 0; i < 16; ++i) { p[i] = __expf(v[i] - mNew); rs += p[i]; }
    rs += __shfl_xor(rs, 16, 32);
    lI = lI * corr + rs;
    mI = mNew;

    // ---- P packs directly into the WMMA A fragment
    h8 plo, phi;
#pragma unroll
    for (int i = 0; i < 8; ++i) { plo[i] = (_Float16)p[i]; phi[i] = (_Float16)p[8 + i]; }
    v16h pf = cat16(plo, phi);

    // ---- broadcast per-row corr to C-layout lanes via per-wave LDS
    Crw[wid][mrow] = corr;
    v4f ca = *(const v4f*)&Crw[wid][hb8];
    v4f cb = *(const v4f*)&Crw[wid][hb8 + 4];
#pragma unroll
    for (int nt = 0; nt < 4; ++nt)
#pragma unroll
      for (int j = 0; j < 8; ++j)
        ao[nt][j] *= (j < 4) ? ca[j] : cb[j - 4];

    // ---- O += P(16x32) @ V(32x64)
#pragma unroll
    for (int nt = 0; nt < 4; ++nt) {
      int ecol = nt * 16 + mrow;
      v16h vfrag = cat16(*(const h8*)&Vst[cur][ecol][hb8],
                         *(const h8*)&Vst[cur][ecol][hb8 + 16]);
      ao[nt] = wmma_f16(pf, vfrag, ao[nt]);
    }

    if (more) {
      *(h8*)&Ks[nxt][ksl][ke8] = kn;
      *(h8*)&Vst[nxt][ve][vs8] = vn;
    }
    __syncthreads();
  }

  // ---- normalize: broadcast lI per row, store f16 [b, l, h, e]
  Crw[wid][mrow] = lI;
  v4f la = *(const v4f*)&Crw[wid][hb8];
  v4f lb = *(const v4f*)&Crw[wid][hb8 + 4];
  size_t obase = ((size_t)(b * AT_L + l0)) * AT_HD + h * 64;
#pragma unroll
  for (int nt = 0; nt < 4; ++nt) {
    int e = nt * 16 + mrow;
#pragma unroll
    for (int j = 0; j < 8; ++j) {
      float denom = (j < 4) ? la[j] : lb[j - 4];
      O[obase + (size_t)(j + hb8) * AT_HD + e] = (_Float16)(ao[nt][j] / denom);
    }
  }
}

// ---------------------------------------------------------------------------
extern "C" void kernel_launch(void* const* d_in, const int* in_sizes, int n_in,
                              void* d_out, int out_size, void* d_ws, size_t ws_size,
                              hipStream_t stream) {
  (void)in_sizes; (void)n_in; (void)out_size; (void)ws_size;

  const float* tgt   = (const float*)d_in[0];
  const float* src   = (const float*)d_in[1];
  const float* val   = (const float*)d_in[2];
  const float* Wq    = (const float*)d_in[3];
  const float* bq    = (const float*)d_in[4];
  const float* Wk    = (const float*)d_in[5];
  const float* bk    = (const float*)d_in[6];
  const float* Wv    = (const float*)d_in[7];
  const float* bv    = (const float*)d_in[8];
  const float* Wo    = (const float*)d_in[9];
  const float* bo    = (const float*)d_in[10];
  const float* alpha = (const float*)d_in[11];
  const float* beta  = (const float*)d_in[12];
  float* out = (float*)d_out;

  // workspace carve-up (all f16)
  _Float16* qbuf = (_Float16*)d_ws;                        // 8192*1024
  _Float16* kbuf = qbuf + (size_t)8192 * 1024;             // 1000*1024
  _Float16* vbuf = kbuf + (size_t)1000 * 1024;             // 1000*1024
  _Float16* attn = vbuf + (size_t)1000 * 1024;             // 8192*1024
  _Float16* WTq  = attn + (size_t)8192 * 1024;             // [1024][1024]
  _Float16* WTk  = WTq  + (size_t)1024 * 1024;             // [1024][4096]
  _Float16* WTv  = WTk  + (size_t)1024 * 4096;             // [1024][4096]
  _Float16* WTo  = WTv  + (size_t)1024 * 4096;             // [4096][1024]

  dim3 blk(256);

  // one-time weight transpose+convert to f16
  transpose_w_f16<1024, 1024><<<1024 * 1024 / 8 / 256, blk, 0, stream>>>(Wq, WTq);
  transpose_w_f16<4096, 1024><<<4096 * 1024 / 8 / 256, blk, 0, stream>>>(Wk, WTk);
  transpose_w_f16<4096, 1024><<<4096 * 1024 / 8 / 256, blk, 0, stream>>>(Wv, WTv);
  transpose_w_f16<1024, 4096><<<1024 * 4096 / 8 / 256, blk, 0, stream>>>(Wo, WTo);

  // Q = (tgt @ Wq + bq) * (alpha*beta/sqrt(64)) -> f16
  gemm_bias_wmma<1024, 1024, false, true>
      <<<dim3(8192 / GB_BM, 1024 / GB_BN), blk, 0, stream>>>(
      tgt, WTq, bq, qbuf, 8192, alpha, beta, 0.125f);
  // K = src @ Wk + bk -> f16
  gemm_bias_wmma<1024, 4096, false, true>
      <<<dim3(16, 1024 / GB_BN), blk, 0, stream>>>(
      src, WTk, bk, kbuf, 1000, nullptr, nullptr, 1.0f);
  // V = val @ Wv + bv -> f16
  gemm_bias_wmma<1024, 4096, false, true>
      <<<dim3(16, 1024 / GB_BN), blk, 0, stream>>>(
      val, WTv, bv, vbuf, 1000, nullptr, nullptr, 1.0f);
  // attn = softmax(Q K^T) V per (b,h) -> f16
  attn_wmma<<<dim3(8 * 16, 1024 / 128), blk, 0, stream>>>(qbuf, kbuf, vbuf, attn);
  // out = attn @ Wo + bo -> f32
  gemm_bias_wmma<4096, 1024, true, false>
      <<<dim3(8192 / GB_BM, 4096 / GB_BN), blk, 0, stream>>>(
      attn, WTo, bo, out, 8192, nullptr, nullptr, 1.0f);
}
